// Head_63316407877951
// MI455X (gfx1250) — matrix operand order repaired and checked by reading, not compiled, not run on hardware
//
#include <hip/hip_runtime.h>
#include <hip/hip_bf16.h>

#define BATCH  8
#define SEQ    4096
#define DMODEL 768
#define HEAD   64

typedef __attribute__((ext_vector_type(16))) _Float16 v16h;
typedef __attribute__((ext_vector_type(8)))  _Float16 v8h;
typedef __attribute__((ext_vector_type(8)))  float    v8f;
typedef __attribute__((ext_vector_type(4)))  float    v4f;
typedef __attribute__((ext_vector_type(4)))  unsigned int u32x4;
typedef __attribute__((ext_vector_type(8)))  int      i32x8;
typedef __attribute__((ext_vector_type(4)))  int      i32x4;

static __device__ __forceinline__ v8f wmma_f16(v16h a, v16h b, v8f c) {
  // D = A(16x32 f16) * B(32x16 f16) + C(16x16 f32)
  return __builtin_amdgcn_wmma_f32_16x16x32_f16(false, a, false, b, (short)0, c,
                                                false, false);
}

static __device__ __forceinline__ v16h pack16(v8h lo, v8h hi) {
  v16h r;
#pragma unroll
  for (int t = 0; t < 8; ++t) { r[t] = lo[t]; r[t + 8] = hi[t]; }
  return r;
}

// Low 32 bits of a generic pointer to an LDS object = byte offset in LDS.
static __device__ __forceinline__ unsigned lds_off(const void* p) {
  return (unsigned)(unsigned long long)(uintptr_t)p;
}

// TDM: DMA a 2-D tile (f16 elements) from global into LDS.
// Tensor is [tensor_d1 rows x tensor_d0 cols] with row stride stride0 (elems),
// tile is [tile_d1 rows x tile_d0 cols] starting at `gaddr`. D# layout per
// CDNA5 ISA ch.8: group0 {count=1, lds_addr, global_addr, type=2},
// group1 {mask=0, data_size=1(2B), dims/strides}; groups 2-3 unused (2-D).
static __device__ __forceinline__ void tdm_load_2d(
    unsigned lds_addr, const void* gaddr, unsigned tensor_d0,
    unsigned tensor_d1, unsigned long long stride0, unsigned tile_d0,
    unsigned tile_d1) {
  unsigned long long ga = (unsigned long long)(uintptr_t)gaddr;
  u32x4 g0;
  g0[0] = 1u;                                   // count=1 (valid), user mode
  g0[1] = lds_addr;                             // LDS byte address
  g0[2] = (unsigned)ga;                         // global_addr[31:0]
  g0[3] = (unsigned)(ga >> 32) | (2u << 30);    // global_addr[56:32] | type=2
  i32x8 g1;
  g1[0] = 0x00010000;                           // wg_mask=0, data_size=2B
  g1[1] = (int)(tensor_d0 << 16);               // barrier_addr=0 | dim0[15:0]
  g1[2] = (int)((tensor_d0 >> 16) | (tensor_d1 << 16));  // dim0 hi | dim1 lo
  g1[3] = (int)((tensor_d1 >> 16) | (tile_d0 << 16));    // dim1 hi | tile_d0
  g1[4] = (int)tile_d1;                         // tile_d1 | tile_d2=0
  g1[5] = (int)(unsigned)stride0;               // dim0_stride[31:0]
  g1[6] = (int)((stride0 >> 32) & 0xFFFF);      // dim0_stride[47:32]
  g1[7] = 0;
  // 6-arg (clang-23/therock) form: groups 2,3 + spare group + cpol.
  __builtin_amdgcn_tensor_load_to_lds(g0, g1, (i32x4)0, (i32x4)0, (i32x8)0, 0);
}

// ---------------------------------------------------------------------------
// Kernel 1: transpose + f16-convert weights: W[768][64] f32 -> wT[3][64][768]
// ---------------------------------------------------------------------------
__global__ void transpose_weights(const float* __restrict__ Wk,
                                  const float* __restrict__ Wq,
                                  const float* __restrict__ Wv,
                                  _Float16* __restrict__ wT) {
  int idx = blockIdx.x * blockDim.x + threadIdx.x;   // [3][HEAD][DMODEL]
  if (idx >= 3 * HEAD * DMODEL) return;
  int m = idx / (HEAD * DMODEL);
  int r = idx % (HEAD * DMODEL);
  int n = r / DMODEL;
  int k = r % DMODEL;
  const float* W = (m == 0) ? Wk : ((m == 1) ? Wq : Wv);
  wT[idx] = (_Float16)W[k * HEAD + n];
}

// ---------------------------------------------------------------------------
// Kernel 2: fused QKV projection. One wave computes a 16x64 tile of q, k, v.
//   qh, kh: [B][S][64] f16 row-major.  vT: [B][64][S] f16 (transposed).
// ---------------------------------------------------------------------------
__global__ __launch_bounds__(128) void proj_qkv(
    const float* __restrict__ ix, const _Float16* __restrict__ wT,
    _Float16* __restrict__ qh, _Float16* __restrict__ kh,
    _Float16* __restrict__ vT) {
  const int lane = threadIdx.x & 31;
  const int wave = threadIdx.x >> 5;
  const int tile = blockIdx.x * 4 + wave;      // 0 .. B*S/16-1 (exact, no guard)
  const int b    = tile / (SEQ / 16);
  const int row0 = (tile % (SEQ / 16)) * 16;
  const int lrow = lane & 15;
  const bool hi  = lane >= 16;

  v8f acc[3][4];
#pragma unroll
  for (int m = 0; m < 3; ++m)
#pragma unroll
    for (int nt = 0; nt < 4; ++nt) acc[m][nt] = (v8f)0.f;

  const float* arow = ix + (size_t)(b * SEQ + row0 + lrow) * DMODEL;

  for (int kc = 0; kc < DMODEL; kc += 32) {
    // A fragment (16x32 f16): lanes<16 hold K kc+0..7 & kc+16..23 of row lrow,
    // lanes>=16 hold K kc+8..15 & kc+24..31.
    const int ka = kc + (hi ? 8 : 0);
    v4f f0 = *(const v4f*)(arow + ka);
    v4f f1 = *(const v4f*)(arow + ka + 4);
    v4f f2 = *(const v4f*)(arow + ka + 16);
    v4f f3 = *(const v4f*)(arow + ka + 20);
    v16h a;
#pragma unroll
    for (int t = 0; t < 4; ++t) {
      a[t]      = (_Float16)f0[t];
      a[t + 4]  = (_Float16)f1[t];
      a[t + 8]  = (_Float16)f2[t];
      a[t + 12] = (_Float16)f3[t];
    }

    const int kb = kc + (hi ? 8 : 0);
#pragma unroll
    for (int m = 0; m < 3; ++m) {
      const _Float16* wbase = wT + (size_t)m * HEAD * DMODEL + kb;
#pragma unroll
      for (int nt = 0; nt < 4; ++nt) {
        // B fragment (32x16): lane holds column nt*16+lrow, K kb+0..7 & kb+16..23.
        const _Float16* wp = wbase + (size_t)(nt * 16 + lrow) * DMODEL;
        v16h bf = pack16(*(const v8h*)wp, *(const v8h*)(wp + 16));
        acc[m][nt] = wmma_f16(a, bf, acc[m][nt]);
      }
    }
  }

  // C layout: element v of v8f is row (row0 + v + 8*hi), column lrow (+16*nt).
#pragma unroll
  for (int v = 0; v < 8; ++v) {
    const int row = row0 + v + (hi ? 8 : 0);
#pragma unroll
    for (int nt = 0; nt < 4; ++nt) {
      const int n = nt * 16 + lrow;
      kh[(size_t)(b * SEQ + row) * HEAD + n] = (_Float16)acc[0][nt][v];
      qh[(size_t)(b * SEQ + row) * HEAD + n] = (_Float16)acc[1][nt][v];
      vT[((size_t)b * HEAD + n) * SEQ + row] = (_Float16)acc[2][nt][v];
    }
  }
}

// ---------------------------------------------------------------------------
// Kernel 3: flash attention. 4 waves/block share one batch's K/V stream:
// TDM double-buffers 32-key K/V tiles into LDS (TENSORcnt-pipelined), each
// wave runs 4 WMMAs for Q*K^T, f32 online softmax, 4 WMMAs for P*V.
// ---------------------------------------------------------------------------
__global__ __launch_bounds__(128) void attn(
    const _Float16* __restrict__ qh, const _Float16* __restrict__ kh,
    const _Float16* __restrict__ vT, float* __restrict__ out) {
  __shared__ __align__(16) _Float16 kbuf[2][32 * HEAD];  // [key][h]   4KB x2
  __shared__ __align__(16) _Float16 vbuf[2][HEAD * 32];  // [h][key]   4KB x2
  __shared__ __align__(16) _Float16 pbuf[4][16 * 32];    // P transpose / wave

  const int lane = threadIdx.x & 31;
  const int wave = threadIdx.x >> 5;
  const int tile = blockIdx.x * 4 + wave;
  const int b    = tile / (SEQ / 16);        // same b for all 4 waves
  const int row0 = (tile % (SEQ / 16)) * 16;
  const int lrow = lane & 15;
  const bool hi  = lane >= 16;

  const _Float16* kbase = kh + (size_t)b * SEQ * HEAD;
  const _Float16* vbase = vT + (size_t)b * HEAD * SEQ;

  // Q A-fragments for K-dim 0..31 and 32..63 (resident for whole loop).
  v16h aq[2];
  const _Float16* qrow = qh + (size_t)(b * SEQ + row0 + lrow) * HEAD;
#pragma unroll
  for (int kk = 0; kk < 2; ++kk) {
    const int ka = kk * 32 + (hi ? 16 : 0);
    aq[kk] = pack16(*(const v8h*)(qrow + ka), *(const v8h*)(qrow + ka + 8));
  }

  float mrow[8], lsum[8];
#pragma unroll
  for (int v = 0; v < 8; ++v) { mrow[v] = -3.0e38f; lsum[v] = 0.f; }
  v8f o[4];
#pragma unroll
  for (int nt = 0; nt < 4; ++nt) o[nt] = (v8f)0.f;

  _Float16* pw = pbuf[wave];

  // Prime the pipeline: tile j=0 into buffer 0 (wave 0 drives the TDM).
  if (wave == 0) {
    tdm_load_2d(lds_off(&kbuf[0][0]), kbase, HEAD, SEQ, HEAD, HEAD, 32);
    tdm_load_2d(lds_off(&vbuf[0][0]), vbase, SEQ, HEAD, SEQ, 32, HEAD);
  }

  for (int it = 0; it < SEQ / 32; ++it) {
    const int j   = it * 32;
    const int cur = it & 1;
    if (wave == 0) __builtin_amdgcn_s_wait_tensorcnt(0);  // buf[cur] landed
    __syncthreads();  // all waves: prev tile consumed AND buf[cur] visible
    if (wave == 0 && it + 1 < SEQ / 32) {   // prefetch next tile via TDM,
      const int jn = j + 32;                // overlapped with compute below
      tdm_load_2d(lds_off(&kbuf[cur ^ 1][0]), kbase + (size_t)jn * HEAD,
                  HEAD, SEQ, HEAD, HEAD, 32);
      tdm_load_2d(lds_off(&vbuf[cur ^ 1][0]), vbase + jn,
                  SEQ, HEAD, SEQ, 32, HEAD);
    }

    // scores S = Q * K^T for 32 keys (two 16-key C fragments) from LDS
    v8f c[2];
#pragma unroll
    for (int jt = 0; jt < 2; ++jt) {
      const _Float16* krow = &kbuf[cur][(jt * 16 + lrow) * HEAD];
      v16h b0 = *(const v16h*)(krow + (hi ? 16 : 0));        // K-dim 0..31
      v16h b1 = *(const v16h*)(krow + 32 + (hi ? 16 : 0));   // K-dim 32..63
      v8f cc = (v8f)0.f;
      cc = wmma_f16(aq[0], b0, cc);
      cc = wmma_f16(aq[1], b1, cc);
      c[jt] = cc;
    }

    // online softmax; rows of a C tile live across 16-lane halves
    float alpha[8];
#pragma unroll
    for (int v = 0; v < 8; ++v) {
      float s0 = c[0][v] * 0.125f;   // 1/sqrt(64)
      float s1 = c[1][v] * 0.125f;
      float t = fmaxf(s0, s1);
      t = fmaxf(t, __shfl_xor(t, 1, 32));
      t = fmaxf(t, __shfl_xor(t, 2, 32));
      t = fmaxf(t, __shfl_xor(t, 4, 32));
      t = fmaxf(t, __shfl_xor(t, 8, 32));
      const float mn = fmaxf(mrow[v], t);
      alpha[v] = __expf(mrow[v] - mn);
      mrow[v]  = mn;
      float p0 = __expf(s0 - mn);
      float p1 = __expf(s1 - mn);
      float ts = p0 + p1;
      ts += __shfl_xor(ts, 1, 32);
      ts += __shfl_xor(ts, 2, 32);
      ts += __shfl_xor(ts, 4, 32);
      ts += __shfl_xor(ts, 8, 32);
      lsum[v] = lsum[v] * alpha[v] + ts;
      // stash P (f16) row-major 16x32 for the C->A transpose
      const int r = v + (hi ? 8 : 0);
      pw[r * 32 + lrow]      = (_Float16)p0;
      pw[r * 32 + lrow + 16] = (_Float16)p1;
    }
#pragma unroll
    for (int nt = 0; nt < 4; ++nt)
#pragma unroll
      for (int v = 0; v < 8; ++v) o[nt][v] *= alpha[v];

    // reload P as an A fragment (same-wave LDS: DS ordering + waits)
    const _Float16* pr = pw + lrow * 32 + (hi ? 16 : 0);
    v16h ap = pack16(*(const v8h*)pr, *(const v8h*)(pr + 8));

    // O += P * V from LDS (V transposed => contiguous per-lane B loads)
#pragma unroll
    for (int nt = 0; nt < 4; ++nt) {
      const _Float16* vrow = &vbuf[cur][(nt * 16 + lrow) * 32];
      v16h bv = *(const v16h*)(vrow + (hi ? 16 : 0));
      o[nt] = wmma_f16(ap, bv, o[nt]);
    }
  }

  // normalize and store f32 output [B][S][64]
#pragma unroll
  for (int v = 0; v < 8; ++v) {
    const float inv = 1.0f / lsum[v];
    const int row = row0 + v + (hi ? 8 : 0);
#pragma unroll
    for (int nt = 0; nt < 4; ++nt)
      out[(size_t)(b * SEQ + row) * HEAD + nt * 16 + lrow] = o[nt][v] * inv;
  }
}

// ---------------------------------------------------------------------------
extern "C" void kernel_launch(void* const* d_in, const int* in_sizes, int n_in,
                              void* d_out, int out_size, void* d_ws,
                              size_t ws_size, hipStream_t stream) {
  const float* ix = (const float*)d_in[0];
  const float* Wk = (const float*)d_in[1];
  const float* Wq = (const float*)d_in[2];
  const float* Wv = (const float*)d_in[3];
  float* out = (float*)d_out;

  const size_t QKV = (size_t)BATCH * SEQ * HEAD;   // 2M f16 elements each
  _Float16* qh = (_Float16*)d_ws;
  _Float16* kh = qh + QKV;
  _Float16* vT = kh + QKV;
  _Float16* wT = vT + QKV;                         // 3*64*768 f16

  transpose_weights<<<dim3((3 * HEAD * DMODEL + 255) / 256), dim3(256), 0,
                      stream>>>(Wk, Wq, Wv, wT);

  const int blocks = (BATCH * SEQ / 16) / 4;       // 512 blocks, 4 waves each
  proj_qkv<<<dim3(blocks), dim3(128), 0, stream>>>(ix, wT, qh, kh, vT);
  attn<<<dim3(blocks), dim3(128), 0, stream>>>(qh, kh, vT, out);
}
